// DKVMN_26577257628345
// MI455X (gfx1250) — compile-verified, matrix operand order checked by reading, not run on hardware
//
#include <hip/hip_runtime.h>

// ---------------- problem constants ----------------
#define B_SZ   512
#define S_SZ   512
#define T_TOK  (B_SZ * S_SZ)      // 262144 tokens
#define K_DIM  128
#define V_DIM  256
#define VM_DIM 256
#define M_SZ   64
#define F_DIM  128
#define CAT_DIM (VM_DIM + K_DIM)  // 384

// ---------------- workspace layout (bytes) ----------------
// w   : f32  [T,64]   =  64 MB   @ 0
// e   : bf16 [T,256]  = 128 MB   @ 64MB
// a   : bf16 [T,256]  = 128 MB   @ 192MB
// k   : bf16 [T,128]  =  64 MB   @ 320MB
// r   : bf16 [T,256]  = 128 MB   @ 384MB
// total 512 MB
#define WS_W_OFF 0
#define WS_E_OFF (67108864ULL)
#define WS_A_OFF (201326592ULL)
#define WS_K_OFF (335544320ULL)
#define WS_R_OFF (402653184ULL)

typedef __attribute__((ext_vector_type(16))) __bf16 v16bf;
typedef __attribute__((ext_vector_type(8)))  float  v8f;

// K-pair offset for the CDNA5 16-bit A/B fragment layout (16x32):
// lanes 0-15: VGPR0..3 -> K {0..7}, VGPR4..7 -> K {16..23}
// lanes16-31: VGPR0..3 -> K {8..15}, VGPR4..7 -> K {24..31}
__device__ __forceinline__ int kpair_off(int i, int lane) {
  return ((lane >= 16) ? 8 : 0) + ((i < 4) ? (2 * i) : (16 + 2 * (i - 4)));
}

__device__ __forceinline__ void load_frag_f32(const float* __restrict__ src, int kbase,
                                              int lane, v16bf& frag) {
#pragma unroll
  for (int i = 0; i < 8; ++i) {
    const float2 p = *(const float2*)(src + kbase + kpair_off(i, lane));
    frag[2 * i]     = (__bf16)p.x;
    frag[2 * i + 1] = (__bf16)p.y;
  }
}

__device__ __forceinline__ void load_frag_bf16(const __bf16* __restrict__ src, int kbase,
                                               int lane, v16bf& frag) {
#pragma unroll
  for (int i = 0; i < 8; ++i) {
    const int kk    = kbase + kpair_off(i, lane);
    frag[2 * i]     = src[kk];
    frag[2 * i + 1] = src[kk + 1];
  }
}

// ---------------- 1) gather k rows -> bf16 ----------------
__global__ void k_gather_k(const int* __restrict__ q, const float* __restrict__ k_emb,
                           __bf16* __restrict__ ws_k) {
  const int tok = blockIdx.x * 2 + (threadIdx.x >> 7);
  const int j   = threadIdx.x & 127;
  const float v = k_emb[(size_t)q[tok] * K_DIM + j];
  ws_k[(size_t)tok * K_DIM + j] = (__bf16)v;
}

// ---------------- 2a) w logits: [T,64] = k[T,128] @ Mk^T ----------------
__global__ void k_gemm_w(const int* __restrict__ q, const float* __restrict__ k_emb,
                         const float* __restrict__ Mk, float* __restrict__ ws_w) {
  const int wave = threadIdx.x >> 5;
  const int lane = threadIdx.x & 31;
  const int tokenTile = blockIdx.x * 2 + (wave >> 2);  // 2 token tiles / block
  const int colTile   = wave & 3;                      // 64 outputs -> 4 col tiles
  const int rowA = tokenTile * 16 + (lane & 15);
  const float* srcA = k_emb + (size_t)q[rowA] * K_DIM;
  const float* srcB = Mk + (size_t)(colTile * 16 + (lane & 15)) * K_DIM;

  v8f acc = {};
#pragma unroll
  for (int ks = 0; ks < K_DIM / 32; ++ks) {
    v16bf af, bf_;
    load_frag_f32(srcA, ks * 32, lane, af);
    load_frag_f32(srcB, ks * 32, lane, bf_);
    acc = __builtin_amdgcn_wmma_f32_16x16x32_bf16(false, af, false, bf_, (short)0, acc,
                                                  false, false);
  }
  const int n = lane & 15;
#pragma unroll
  for (int j = 0; j < 8; ++j) {
    const int m = j + ((lane < 16) ? 0 : 8);
    ws_w[(size_t)(tokenTile * 16 + m) * M_SZ + colTile * 16 + n] = acc[j];
  }
}

// ---------------- 2b) per-token softmax over 64 (in place) ----------------
__global__ void k_softmax64(float* __restrict__ ws_w) {
  const int tok = blockIdx.x * 256 + threadIdx.x;
  float* p = ws_w + (size_t)tok * M_SZ;
  float lg[M_SZ];
  float mx = -3.4e38f;
#pragma unroll
  for (int m = 0; m < M_SZ; ++m) { lg[m] = p[m]; mx = fmaxf(mx, lg[m]); }
  float sum = 0.f;
#pragma unroll
  for (int m = 0; m < M_SZ; ++m) { lg[m] = __expf(lg[m] - mx); sum += lg[m]; }
  const float inv = 1.f / sum;
#pragma unroll
  for (int m = 0; m < M_SZ; ++m) p[m] = lg[m] * inv;
}

// ---------------- 3) e = sigmoid(v@We^T+be), a = tanh(v@Wa^T+ba) ----------------
__global__ void k_gemm_ea(const int* __restrict__ qa, const float* __restrict__ v_emb,
                          const float* __restrict__ We, const float* __restrict__ be,
                          const float* __restrict__ Wa, const float* __restrict__ ba,
                          __bf16* __restrict__ ws_e, __bf16* __restrict__ ws_a) {
  const int wave = threadIdx.x >> 5;
  const int lane = threadIdx.x & 31;
  const int tokenTile = blockIdx.x;
  const int colTile   = blockIdx.y * 8 + wave;   // 0..31
  const int which     = colTile >> 4;            // 0=e, 1=a
  const int colbase   = (colTile & 15) * 16;

  const float* W    = which ? Wa : We;
  const float* bias = which ? ba : be;
  __bf16*      dst  = which ? ws_a : ws_e;

  const int rowA = tokenTile * 16 + (lane & 15);
  const float* srcA = v_emb + (size_t)qa[rowA] * V_DIM;
  const float* srcB = W + (size_t)(colbase + (lane & 15)) * V_DIM;

  v8f acc = {};
#pragma unroll
  for (int ks = 0; ks < V_DIM / 32; ++ks) {
    v16bf af, bf_;
    load_frag_f32(srcA, ks * 32, lane, af);
    load_frag_f32(srcB, ks * 32, lane, bf_);
    acc = __builtin_amdgcn_wmma_f32_16x16x32_bf16(false, af, false, bf_, (short)0, acc,
                                                  false, false);
  }
  const int n = lane & 15;
  const float bs = bias[colbase + n];
#pragma unroll
  for (int j = 0; j < 8; ++j) {
    const int m = j + ((lane < 16) ? 0 : 8);
    const float val = acc[j] + bs;
    const float act = which ? tanhf(val) : (1.f / (1.f + __expf(-val)));
    dst[(size_t)(tokenTile * 16 + m) * VM_DIM + colbase + n] = (__bf16)act;
  }
}

// ---------------- 4) sequential scan over S; state in VGPRs ----------------
__global__ void k_scan(const float* __restrict__ ws_w, const __bf16* __restrict__ ws_e,
                       const __bf16* __restrict__ ws_a, const float* __restrict__ Mv0,
                       __bf16* __restrict__ ws_r) {
  __shared__ float wrow[M_SZ];
  const int b = blockIdx.x;
  const int v = threadIdx.x;          // 256 threads: one state column each

  float Mv[M_SZ];
#pragma unroll
  for (int m = 0; m < M_SZ; ++m) Mv[m] = Mv0[m * VM_DIM + v];

  const float* wptr = ws_w + (size_t)b * S_SZ * M_SZ;
  for (int t = 0; t < S_SZ; ++t) {
    if (threadIdx.x < M_SZ) wrow[threadIdx.x] = wptr[t * M_SZ + threadIdx.x];
    __syncthreads();
    const size_t idx = ((size_t)b * S_SZ + t) * VM_DIM + v;
    const float e = (float)ws_e[idx];
    const float a = (float)ws_a[idx];
    float r = 0.f;
#pragma unroll
    for (int m = 0; m < M_SZ; ++m) {
      const float wm  = wrow[m];
      const float old = Mv[m];
      r     = fmaf(wm, old, r);                 // read BEFORE write
      Mv[m] = fmaf(wm, a - e * old, old);       // Mv += w*(a - e*Mv)
    }
    ws_r[idx] = (__bf16)r;
    __syncthreads();
  }
}

// ---------------- 5) head: f = tanh([r;k]@Wf^T + bf); p = f.Wp + bp ----------------
__global__ void k_final(const __bf16* __restrict__ ws_r, const __bf16* __restrict__ ws_k,
                        const float* __restrict__ Wf, const float* __restrict__ bfv,
                        const float* __restrict__ Wp, const float* __restrict__ bp,
                        float* __restrict__ out) {
  __shared__ float accs[8][16];
  const int wave = threadIdx.x >> 5;   // col tile 0..7 (128 outputs)
  const int lane = threadIdx.x & 31;
  const int tokenTile = blockIdx.x;
  const int colbase   = wave * 16;

  const int rowA = tokenTile * 16 + (lane & 15);
  const __bf16* srcR = ws_r + (size_t)rowA * VM_DIM;
  const __bf16* srcK = ws_k + (size_t)rowA * K_DIM;
  const float*  srcB = Wf + (size_t)(colbase + (lane & 15)) * CAT_DIM;

  v8f acc = {};
#pragma unroll
  for (int ks = 0; ks < CAT_DIM / 32; ++ks) {       // 12 k-steps: 8 over r, 4 over k
    v16bf af, bf_;
    if (ks < 8) load_frag_bf16(srcR, ks * 32, lane, af);
    else        load_frag_bf16(srcK, ks * 32 - VM_DIM, lane, af);
    load_frag_f32(srcB, ks * 32, lane, bf_);
    acc = __builtin_amdgcn_wmma_f32_16x16x32_bf16(false, af, false, bf_, (short)0, acc,
                                                  false, false);
  }
  const int n = lane & 15;
  const float bs = bfv[colbase + n];
  const float wp = Wp[colbase + n];
#pragma unroll
  for (int j = 0; j < 8; ++j) {
    float part = tanhf(acc[j] + bs) * wp;
    // sum over n (16 lanes of each half-wave)
#pragma unroll
    for (int off = 8; off >= 1; off >>= 1) part += __shfl_xor(part, off);
    if (lane == 0)  accs[wave][j]     = part;
    if (lane == 16) accs[wave][j + 8] = part;
  }
  __syncthreads();
  if (threadIdx.x < 16) {
    float s = bp[0];
#pragma unroll
    for (int w = 0; w < 8; ++w) s += accs[w][threadIdx.x];
    out[(size_t)tokenTile * 16 + threadIdx.x] = s;
  }
}

// ---------------- launch ----------------
extern "C" void kernel_launch(void* const* d_in, const int* in_sizes, int n_in,
                              void* d_out, int out_size, void* d_ws, size_t ws_size,
                              hipStream_t stream) {
  const int*   q     = (const int*)d_in[0];
  const int*   qa    = (const int*)d_in[1];
  const float* k_emb = (const float*)d_in[2];
  const float* v_emb = (const float*)d_in[3];
  const float* Wf    = (const float*)d_in[4];
  const float* bfv   = (const float*)d_in[5];
  const float* Wp    = (const float*)d_in[6];
  const float* bp    = (const float*)d_in[7];
  const float* We    = (const float*)d_in[8];
  const float* be    = (const float*)d_in[9];
  const float* Wa    = (const float*)d_in[10];
  const float* ba    = (const float*)d_in[11];
  const float* Mk    = (const float*)d_in[12];
  const float* Mv0   = (const float*)d_in[13];
  float* out = (float*)d_out;

  char* ws = (char*)d_ws;
  float*  ws_w = (float*)(ws + WS_W_OFF);
  __bf16* ws_e = (__bf16*)(ws + WS_E_OFF);
  __bf16* ws_a = (__bf16*)(ws + WS_A_OFF);
  __bf16* ws_k = (__bf16*)(ws + WS_K_OFF);
  __bf16* ws_r = (__bf16*)(ws + WS_R_OFF);

  k_gather_k<<<T_TOK / 2, 256, 0, stream>>>(q, k_emb, ws_k);
  k_gemm_w<<<T_TOK / 32, 256, 0, stream>>>(q, k_emb, Mk, ws_w);
  k_softmax64<<<T_TOK / 256, 256, 0, stream>>>(ws_w);
  k_gemm_ea<<<dim3(T_TOK / 16, 4), 256, 0, stream>>>(qa, v_emb, We, be, Wa, ba, ws_e, ws_a);
  k_scan<<<B_SZ, 256, 0, stream>>>(ws_w, ws_e, ws_a, Mv0, ws_r);
  k_final<<<T_TOK / 16, 256, 0, stream>>>(ws_r, ws_k, Wf, bfv, Wp, bp, out);
}
